// Qwen3MoEExperts_12360915878734
// MI455X (gfx1250) — matrix operand order repaired and checked by reading, not compiled, not run on hardware
//
#include <hip/hip_runtime.h>

// ---------------- problem constants ----------------
constexpr int cT = 1024;   // tokens
constexpr int cH = 2048;   // hidden
constexpr int cI = 768;    // intermediate
constexpr int cE = 32;     // experts
constexpr int cTK = cT * 4;    // 4096 grouped rows (top-k = 4)

// ---------------- vector types ----------------
typedef __attribute__((ext_vector_type(16))) __bf16 bf16x16;
typedef __attribute__((ext_vector_type(2)))  __bf16 bf16x2;
typedef __attribute__((ext_vector_type(8)))  float  f32x8;
typedef __attribute__((ext_vector_type(4)))  float  f32x4;
typedef __attribute__((ext_vector_type(4)))  unsigned int u32x4;

union FragB16 { bf16x16 v; u32x4 q[2]; };

__device__ inline unsigned short f32_to_bf16_bits(float f) {
    unsigned int u = __builtin_bit_cast(unsigned int, f);
    unsigned int r = u + 0x7FFFu + ((u >> 16) & 1u);   // round-to-nearest-even
    return (unsigned short)(r >> 16);
}

#if __has_builtin(__builtin_amdgcn_cvt_pk_bf16_f32)
__device__ inline unsigned int pack_bf16(float a, float b) {
    bf16x2 h = __builtin_amdgcn_cvt_pk_bf16_f32(a, b);
    return __builtin_bit_cast(unsigned int, h);
}
#else
__device__ inline unsigned int pack_bf16(float a, float b) {
    return (unsigned int)f32_to_bf16_bits(a) |
           ((unsigned int)f32_to_bf16_bits(b) << 16);
}
#endif

// LDS tile row stride (bf16 elems): 32 data + 8 pad -> 80B rows, 16B aligned,
// conflict-free across 16 rows (20 dwords/row, 20*r mod 64 all distinct).
constexpr int LSTR = 40;

// ---------------- small utility kernels ----------------
__global__ void zero_f32(float* __restrict__ p, int n) {
    int i = blockIdx.x * blockDim.x + threadIdx.x;
    if (i < n) p[i] = 0.0f;
}

__global__ void zero_i32(int* __restrict__ p, int n) {
    int i = threadIdx.x;
    if (i < n) p[i] = 0;
}

__global__ void cvt_x_bf16(const float* __restrict__ x, unsigned short* __restrict__ Xbf,
                           int n8) {
    int i = blockIdx.x * blockDim.x + threadIdx.x;   // one 8-elem chunk per thread
    if (i < n8) {
        const f32x4 a = *(const f32x4*)(x + (size_t)i * 8);
        const f32x4 b = *(const f32x4*)(x + (size_t)i * 8 + 4);
        u32x4 o;
        o.x = pack_bf16(a.x, a.y); o.y = pack_bf16(a.z, a.w);
        o.z = pack_bf16(b.x, b.y); o.w = pack_bf16(b.z, b.w);
        *(u32x4*)(Xbf + (size_t)i * 8) = o;
    }
}

__global__ void moe_count(const int* __restrict__ ids, int* __restrict__ counts) {
    int i = blockIdx.x * blockDim.x + threadIdx.x;
    if (i < cTK) atomicAdd(&counts[ids[i]], 1);
}

__global__ void moe_scan(const int* __restrict__ counts, int* __restrict__ offsets,
                         int* __restrict__ cursor) {
    if (threadIdx.x == 0) {
        int acc = 0;
        for (int e = 0; e < cE; ++e) {
            offsets[e] = acc;
            cursor[e]  = acc;
            acc += counts[e];
        }
        offsets[cE] = acc;
    }
}

__global__ void moe_scatter(const int* __restrict__ ids, const float* __restrict__ ew,
                            int* __restrict__ cursor, int* __restrict__ row_token,
                            float* __restrict__ row_weight) {
    int i = blockIdx.x * blockDim.x + threadIdx.x;
    if (i < cTK) {
        int e   = ids[i];
        int pos = atomicAdd(&cursor[e], 1);
        row_token[pos]  = i >> 2;
        row_weight[pos] = ew[i];
    }
}

// ---------------- fused gate/up GEMM:  Hh = silu(X Wg^T) * (X Wu^T) ----------------
// grid: (cI/32, cE). block: 256 = 8 waves. Tile: M=128 x N=32, K-chunks of 32 over cH.
// Double-buffered LDS, software-pipelined global->LDS against WMMA.
// OOB row guards are replaced by index clamping: garbage rows only affect output
// rows that the epilogue discards, so loads are unconditional.
__global__ __launch_bounds__(256) void moe_gemm_gateup(
    const unsigned short* __restrict__ Xbf, const float* __restrict__ Wg,
    const float* __restrict__ Wu, const int* __restrict__ offsets,
    const int* __restrict__ row_token, unsigned short* __restrict__ Hbuf) {

    __shared__ unsigned short Xs[2][128 * LSTR];
    __shared__ unsigned short Gs[2][32 * LSTR];
    __shared__ unsigned short Us[2][32 * LSTR];

    const int e    = blockIdx.y;
    const int n0   = blockIdx.x * 32;
    const int base = offsets[e];
    const int cnt  = offsets[e + 1] - base;
    const int tid  = threadIdx.x;
    const int wave = tid >> 5;
    const int lane = tid & 31;
    const int lrow = lane & 15;
    const int akb  = (lane < 16) ? 0 : 16;   // A-frag K base
    const int bkb  = (lane < 16) ? 0 : 16;   // B-frag K base

    // weight staging: one float4 per thread per matrix per chunk
    const int wm = tid >> 3, wk4 = tid & 7;
    const size_t woff = (size_t)e * cI * cH + (size_t)(n0 + wm) * cH + wk4 * 4;
    const float* gsrc = Wg + woff;
    const float* usrc = Wu + woff;
    const int    wdst = wm * LSTR + wk4 * 4;

    constexpr int NK = cH / 32;   // 64 K-chunks

    for (int row0 = 0; row0 < cnt; row0 += 128) {
        // ---- hoisted gather descriptors: 2 x 16B chunks per thread ----
        const unsigned short* xsrc[2];
        int xdst[2];
        #pragma unroll
        for (int v = 0; v < 2; ++v) {
            int c = tid + v * 256;            // 512 chunks of 8 bf16
            int m = c >> 2, k8 = c & 3;
            int grow = min(row0 + m, cnt - 1);           // clamp: OOB rows discarded later
            int t = row_token[base + grow];
            xsrc[v] = Xbf + (size_t)t * cH + k8 * 8;
            xdst[v] = m * LSTR + k8 * 8;
        }

        u32x4 xr[2];
        f32x4 gr4, ur4;
        auto ldc = [&](int h0) {
            #pragma unroll
            for (int v = 0; v < 2; ++v) xr[v] = *(const u32x4*)(xsrc[v] + h0);
            gr4 = __builtin_nontemporal_load((const f32x4*)(gsrc + h0));
            ur4 = __builtin_nontemporal_load((const f32x4*)(usrc + h0));
        };
        auto stc = [&](int b) {
            #pragma unroll
            for (int v = 0; v < 2; ++v) *(u32x4*)&Xs[b][xdst[v]] = xr[v];
            unsigned int* pg = (unsigned int*)&Gs[b][wdst];
            pg[0] = pack_bf16(gr4.x, gr4.y); pg[1] = pack_bf16(gr4.z, gr4.w);
            unsigned int* pu = (unsigned int*)&Us[b][wdst];
            pu[0] = pack_bf16(ur4.x, ur4.y); pu[1] = pack_bf16(ur4.z, ur4.w);
        };

        f32x8 accG[2] = {f32x8{}, f32x8{}};
        f32x8 accU[2] = {f32x8{}, f32x8{}};

        ldc(0);
        stc(0);
        for (int kc = 0; kc < NK; ++kc) {
            const int cur = kc & 1;
            if (kc + 1 < NK) ldc((kc + 1) * 32);
            __syncthreads();

            FragB16 A;
            const unsigned short* arow = &Xs[cur][(wave * 16 + lrow) * LSTR];
            A.q[0] = *(const u32x4*)(arow + akb);
            A.q[1] = *(const u32x4*)(arow + akb + 8);
            #pragma unroll
            for (int ns = 0; ns < 2; ++ns) {
                FragB16 Bg, Bu;
                const unsigned short* gr = &Gs[cur][(ns * 16 + lrow) * LSTR + bkb];
                Bg.q[0] = *(const u32x4*)(gr);
                Bg.q[1] = *(const u32x4*)(gr + 8);
                const unsigned short* ur = &Us[cur][(ns * 16 + lrow) * LSTR + bkb];
                Bu.q[0] = *(const u32x4*)(ur);
                Bu.q[1] = *(const u32x4*)(ur + 8);
                accG[ns] = __builtin_amdgcn_wmma_f32_16x16x32_bf16(
                    false, A.v, false, Bg.v, (short)0, accG[ns], false, false);
                accU[ns] = __builtin_amdgcn_wmma_f32_16x16x32_bf16(
                    false, A.v, false, Bu.v, (short)0, accU[ns], false, false);
            }
            if (kc + 1 < NK) stc((kc + 1) & 1);
        }

        // ---- epilogue: silu(g) * u -> Hbuf (bf16) ----
        #pragma unroll
        for (int ns = 0; ns < 2; ++ns) {
            #pragma unroll
            for (int r = 0; r < 8; ++r) {
                int M = r + ((lane < 16) ? 0 : 8);
                int grow = row0 + wave * 16 + M;
                if (grow < cnt) {
                    float g  = accG[ns][r];
                    float hv = (g / (1.0f + __expf(-g))) * accU[ns][r];
                    Hbuf[(size_t)(base + grow) * cI + n0 + ns * 16 + lrow] =
                        (unsigned short)(pack_bf16(hv, hv) & 0xFFFFu);
                }
            }
        }
    }
}

// ---------------- down GEMM:  y[t] += w * Hh @ Wd^T ----------------
// grid: (cH/32, cE). block: 256 = 8 waves. Tile: M=128 x N=32, K-chunks of 32 over cI.
__global__ __launch_bounds__(256) void moe_gemm_down(
    const float* __restrict__ Wd, const unsigned short* __restrict__ Hbuf,
    const int* __restrict__ offsets, const int* __restrict__ row_token,
    const float* __restrict__ row_weight, float* __restrict__ y) {

    __shared__ unsigned short Hs[2][128 * LSTR];
    __shared__ unsigned short Ds[2][32 * LSTR];

    const int e    = blockIdx.y;
    const int h0   = blockIdx.x * 32;
    const int base = offsets[e];
    const int cnt  = offsets[e + 1] - base;
    const int tid  = threadIdx.x;
    const int wave = tid >> 5;
    const int lane = tid & 31;
    const int lrow = lane & 15;
    const int akb  = (lane < 16) ? 0 : 16;
    const int bkb  = (lane < 16) ? 0 : 16;

    const int wm = tid >> 3, wk4 = tid & 7;
    const float* dsrc = Wd + (size_t)e * cH * cI + (size_t)(h0 + wm) * cI + wk4 * 4;
    const int    wdst = wm * LSTR + wk4 * 4;

    constexpr int NK = cI / 32;   // 24 K-chunks

    for (int row0 = 0; row0 < cnt; row0 += 128) {
        const unsigned short* hsrc[2];
        int hdst[2];
        #pragma unroll
        for (int v = 0; v < 2; ++v) {
            int c = tid + v * 256;
            int m = c >> 2, k8 = c & 3;
            int grow = min(row0 + m, cnt - 1);           // clamp: OOB rows discarded later
            hsrc[v] = Hbuf + (size_t)(base + grow) * cI + k8 * 8;
            hdst[v] = m * LSTR + k8 * 8;
        }

        u32x4 hr[2];
        f32x4 dr4;
        auto ldc = [&](int i0) {
            #pragma unroll
            for (int v = 0; v < 2; ++v) hr[v] = *(const u32x4*)(hsrc[v] + i0);
            dr4 = __builtin_nontemporal_load((const f32x4*)(dsrc + i0));
        };
        auto stc = [&](int b) {
            #pragma unroll
            for (int v = 0; v < 2; ++v) *(u32x4*)&Hs[b][hdst[v]] = hr[v];
            unsigned int* pd = (unsigned int*)&Ds[b][wdst];
            pd[0] = pack_bf16(dr4.x, dr4.y); pd[1] = pack_bf16(dr4.z, dr4.w);
        };

        f32x8 acc[2] = {f32x8{}, f32x8{}};

        ldc(0);
        stc(0);
        for (int kc = 0; kc < NK; ++kc) {
            const int cur = kc & 1;
            if (kc + 1 < NK) ldc((kc + 1) * 32);
            __syncthreads();

            FragB16 A;
            const unsigned short* arow = &Hs[cur][(wave * 16 + lrow) * LSTR];
            A.q[0] = *(const u32x4*)(arow + akb);
            A.q[1] = *(const u32x4*)(arow + akb + 8);
            #pragma unroll
            for (int ns = 0; ns < 2; ++ns) {
                FragB16 B;
                const unsigned short* dr = &Ds[cur][(ns * 16 + lrow) * LSTR + bkb];
                B.q[0] = *(const u32x4*)(dr);
                B.q[1] = *(const u32x4*)(dr + 8);
                acc[ns] = __builtin_amdgcn_wmma_f32_16x16x32_bf16(
                    false, A.v, false, B.v, (short)0, acc[ns], false, false);
            }
            if (kc + 1 < NK) stc((kc + 1) & 1);
        }

        // ---- epilogue: scale by router weight, atomic combine into y ----
        #pragma unroll
        for (int ns = 0; ns < 2; ++ns) {
            #pragma unroll
            for (int r = 0; r < 8; ++r) {
                int M = r + ((lane < 16) ? 0 : 8);
                int grow = row0 + wave * 16 + M;
                if (grow < cnt) {
                    int   t = row_token[base + grow];
                    float w = row_weight[base + grow];
                    atomicAdd(&y[(size_t)t * cH + h0 + ns * 16 + lrow], w * acc[ns][r]);
                }
            }
        }
    }
}

// ---------------- launcher ----------------
extern "C" void kernel_launch(void* const* d_in, const int* in_sizes, int n_in,
                              void* d_out, int out_size, void* d_ws, size_t ws_size,
                              hipStream_t stream) {
    const float* x   = (const float*)d_in[0];
    const float* Wg  = (const float*)d_in[1];
    const float* Wu  = (const float*)d_in[2];
    const float* Wd  = (const float*)d_in[3];
    const int*   ids = (const int*)d_in[4];
    const float* ew  = (const float*)d_in[5];
    float* y = (float*)d_out;

    char* ws = (char*)d_ws;
    int*   counts     = (int*)(ws);
    int*   offsets    = (int*)(ws + 128);
    int*   cursor     = (int*)(ws + 384);
    int*   row_token  = (int*)(ws + 4096);             // 16 KB
    float* row_weight = (float*)(ws + 4096 + 16384);   // 16 KB
    unsigned short* Xbf  = (unsigned short*)(ws + 65536);           // 4 MB
    unsigned short* Hbuf = (unsigned short*)(ws + 8 * 1024 * 1024); // ~6.3 MB

    zero_f32<<<(cT * cH + 255) / 256, 256, 0, stream>>>(y, cT * cH);
    zero_i32<<<1, 64, 0, stream>>>(counts, cE);
    cvt_x_bf16<<<(cT * cH / 8 + 255) / 256, 256, 0, stream>>>(x, Xbf, cT * cH / 8);
    moe_count<<<cTK / 256, 256, 0, stream>>>(ids, counts);
    moe_scan<<<1, 32, 0, stream>>>(counts, offsets, cursor);
    moe_scatter<<<cTK / 256, 256, 0, stream>>>(ids, ew, cursor, row_token, row_weight);

    moe_gemm_gateup<<<dim3(cI / 32, cE), 256, 0, stream>>>(
        Xbf, Wg, Wu, offsets, row_token, Hbuf);
    moe_gemm_down<<<dim3(cH / 32, cE), 256, 0, stream>>>(
        Wd, Hbuf, offsets, row_token, row_weight, y);
}
